// MultiHeadAttention_18769007083521
// MI455X (gfx1250) — matrix operand order repaired and checked by reading, not compiled, not run on hardware
//
#include <hip/hip_runtime.h>
#include <hip/hip_bf16.h>

// ---------------------------------------------------------------------------
// GPT-2 attention block on gfx1250 (MI455X), bf16 WMMA + f32 accumulate.
// Ping-pong double buffering (no register copies), 32x64 tiles per wave in
// the projections, async-to-LDS staging for the Q tile.
// ---------------------------------------------------------------------------

typedef __bf16 bf16_t;
typedef __attribute__((ext_vector_type(16))) __bf16 v16bf;
typedef __attribute__((ext_vector_type(8)))  float  v8f;
typedef __attribute__((ext_vector_type(4)))  int    v4i;

#define WMMA_BF16(A, B, C) \
  __builtin_amdgcn_wmma_f32_16x16x32_bf16(false, (A), false, (B), (short)0, (C), false, false)

// Problem constants (fixed by the reference).
constexpr int B_   = 4;
constexpr int S_   = 1024;
constexpr int D_   = 1024;
constexpr int H_   = 16;
constexpr int DEP_ = 64;
constexpr int PAST = 1024;
constexpr int NS   = 2048;   // PAST + S
constexpr int M_   = B_ * S_;       // 4096 rows for the projections
constexpr int N1_  = 3 * D_;        // 3072

union FragU { v16bf v; uint4 u[2]; };

// A-matrix 16x32 bf16 fragment, row-major source, stride lda (elements).
// Lanes 0-15 = rows; elems 0..7 -> K=8*half+(0..7), elems 8..15 -> K=16+8*half+(0..7).
__device__ inline v16bf load_frag_a(const bf16_t* __restrict__ base, int lda) {
  const int lane = threadIdx.x & 31;
  const int half = lane >> 4;
  const int row  = lane & 15;
  const bf16_t* p = base + (size_t)row * lda + 8 * half;
  FragU f;
  f.u[0] = *reinterpret_cast<const uint4*>(p);
  f.u[1] = *reinterpret_cast<const uint4*>(p + 16);
  return f.v;
}

// B-matrix 32x16 bf16 fragment from TRANSPOSED storage Bt[n][k], stride ldt.
// col = lane&15; elems 0..7 -> K=8*half+(0..7), elems 8..15 -> K=16+8*half+(0..7).
__device__ inline v16bf load_frag_bT(const bf16_t* __restrict__ baseT, int ldt) {
  const int lane  = threadIdx.x & 31;
  const int khalf = lane >> 4;
  const int col   = lane & 15;
  const bf16_t* p = baseT + (size_t)col * ldt + 8 * khalf;
  FragU f;
  f.u[0] = *reinterpret_cast<const uint4*>(p);
  f.u[1] = *reinterpret_cast<const uint4*>(p + 16);
  return f.v;
}

// ---------------------------------------------------------------------------
// Prep kernels
// ---------------------------------------------------------------------------

__global__ void cvt_f32_bf16(const float* __restrict__ s, bf16_t* __restrict__ d) {
  size_t i = (size_t)blockIdx.x * 256 + threadIdx.x;
  d[i] = (bf16_t)s[i];
}

// src: K x N row-major f32; dst: N x K row-major bf16 (transpose + convert).
__global__ void transpose_bf16(const float* __restrict__ src, bf16_t* __restrict__ dst,
                               int K, int N) {
  __shared__ float tile[16][17];
  int n = blockIdx.x * 16 + threadIdx.x;
  int k = blockIdx.y * 16 + threadIdx.y;
  tile[threadIdx.y][threadIdx.x] = src[(size_t)k * N + n];
  __syncthreads();
  int n2 = blockIdx.x * 16 + threadIdx.y;
  int k2 = blockIdx.y * 16 + threadIdx.x;
  dst[(size_t)n2 * K + k2] = (bf16_t)tile[threadIdx.x][threadIdx.y];
}

// past (B,2,H,PAST,DEP) f32 -> Kcat[b,h,pos,dep] bf16 and VcatT[b,h,dep,pos] bf16.
__global__ void past_to_cache(const float* __restrict__ past,
                              bf16_t* __restrict__ Kcat, bf16_t* __restrict__ VcatT) {
  size_t i = (size_t)blockIdx.x * 256 + threadIdx.x;
  int dep = (int)(i & 63);
  size_t r = i >> 6;
  int p  = (int)(r & 1023); r >>= 10;
  int h  = (int)(r & 15);   r >>= 4;
  int kv = (int)(r & 1);
  int b  = (int)(r >> 1);
  float v = past[i];
  size_t bh = (size_t)(b * H_ + h);
  if (kv == 0) Kcat [(bh * NS + p) * DEP_ + dep]  = (bf16_t)v;
  else         VcatT[(bh * DEP_ + dep) * NS + p]  = (bf16_t)v;
}

// ---------------------------------------------------------------------------
// Kernel 1: QKV projection. One wave computes a 32x64 tile of C = Xbf @ W1,
// ping-pong double-buffered fragments (no register copies in steady state).
// ---------------------------------------------------------------------------

__global__ __launch_bounds__(32) void qkv_gemm(
    const bf16_t* __restrict__ Xbf, const bf16_t* __restrict__ W1T,
    const float* __restrict__ b1,
    bf16_t* __restrict__ Qbf, bf16_t* __restrict__ Kcat, bf16_t* __restrict__ VcatT,
    float* __restrict__ present) {
  const int n_base = blockIdx.x * 64;
  const int m_base = blockIdx.y * 32;
  const int lane = threadIdx.x & 31;
  const int half = lane >> 4, col = lane & 15;

  const bf16_t* Ar0 = Xbf + (size_t)m_base * D_;
  const bf16_t* Ar1 = Xbf + (size_t)(m_base + 16) * D_;
  const bf16_t* Bb  = W1T + (size_t)n_base * D_;

  v8f acc[2][4] = {};

  auto load_set = [&](int kn, v16bf& x0, v16bf& x1, v16bf* bb) {
    x0 = load_frag_a(Ar0 + kn, D_);
    x1 = load_frag_a(Ar1 + kn, D_);
#pragma unroll
    for (int t = 0; t < 4; ++t)
      bb[t] = load_frag_bT(Bb + (size_t)t * 16 * D_ + kn, D_);
  };
  auto compute = [&](const v16bf& x0, const v16bf& x1, const v16bf* bb) {
#pragma unroll
    for (int t = 0; t < 4; ++t) {
      acc[0][t] = WMMA_BF16(x0, bb[t], acc[0][t]);
      acc[1][t] = WMMA_BF16(x1, bb[t], acc[1][t]);
    }
  };

  v16bf a0A, a1A, bA[4], a0B, a1B, bB[4];
  load_set(0, a0A, a1A, bA);
#pragma unroll 1
  for (int ki = 0; ki < 15; ++ki) {            // 2 k-steps per iteration
    load_set((2 * ki + 1) * 32, a0B, a1B, bB);
    compute(a0A, a1A, bA);
    load_set((2 * ki + 2) * 32, a0A, a1A, bA);
    compute(a0B, a1B, bB);
  }
  load_set(31 * 32, a0B, a1B, bB);
  compute(a0A, a1A, bA);
  compute(a0B, a1B, bB);

  // Epilogue. Each 64-wide N tile lies entirely inside one of q/k/v and one head.
  const int part = n_base >> 10;
  const int head = (n_base & 1023) >> 6;
#pragma unroll
  for (int r = 0; r < 2; ++r) {
#pragma unroll
    for (int t = 0; t < 4; ++t) {
      const int n = n_base + 16 * t + col;
      const float bias = b1[n];
      const int dep = 16 * t + col;
#pragma unroll
      for (int e = 0; e < 8; ++e) {
        const int m = m_base + 16 * r + e + 8 * half;
        const int b = m >> 10, s = m & 1023;
        const float val = acc[r][t][e] + bias;
        const size_t bh = (size_t)(b * H_ + head);
        if (part == 0) {
          Qbf[(bh * S_ + s) * DEP_ + dep] = (bf16_t)val;
        } else if (part == 1) {
          present[((((size_t)b * 2 + 0) * H_ + head) * S_ + s) * DEP_ + dep] = val;
          Kcat[(bh * NS + PAST + s) * DEP_ + dep] = (bf16_t)val;
        } else {
          present[((((size_t)b * 2 + 1) * H_ + head) * S_ + s) * DEP_ + dep] = val;
          VcatT[(bh * DEP_ + dep) * NS + PAST + s] = (bf16_t)val;
        }
      }
    }
  }
}

// ---------------------------------------------------------------------------
// Kernel 2: fused attention for one (b, h, 16-query tile). 8 waves / block.
// LDS: bf16 probs 16x2048 (64KB) | f32 logits 16x2048 (128KB, reused for AV
// partial reduction) | reduction scratch. Q tile staged via async-to-LDS.
// ---------------------------------------------------------------------------

constexpr size_t SM_PROBS  = (size_t)16 * NS * sizeof(bf16_t);   // 65536
constexpr size_t SM_LOGITS = (size_t)16 * NS * sizeof(float);    // 131072
constexpr size_t SM_RED    = 256 * sizeof(float);                // 1024
constexpr size_t SM_STAT   = 32 * sizeof(float);                 // 128
constexpr size_t SM_TOTAL  = SM_PROBS + SM_LOGITS + SM_RED + SM_STAT;

__global__ __launch_bounds__(256) void attention_kernel(
    const bf16_t* __restrict__ Qbf, const bf16_t* __restrict__ Kcat,
    const bf16_t* __restrict__ VcatT,
    float* __restrict__ attw, bf16_t* __restrict__ AOut) {
  extern __shared__ char smem[];
  bf16_t* probs  = (bf16_t*)smem;
  float*  logits = (float*)(smem + SM_PROBS);
  float*  red    = (float*)(smem + SM_PROBS + SM_LOGITS);
  float*  rstat  = red + 256;
  __shared__ __align__(16) bf16_t qtile[16 * DEP_];   // static LDS, 2KB

  const int b = blockIdx.z, h = blockIdx.y, q_base = blockIdx.x * 16;
  const int tid = threadIdx.x, wave = tid >> 5, lane = tid & 31;
  const int half = lane >> 4, col = lane & 15;
  const size_t bh = (size_t)(b * H_ + h);

  const bf16_t* Q  = Qbf   + (bh * S_ + q_base) * DEP_;  // 16 x 64 row-major
  const bf16_t* K  = Kcat  + bh * NS * DEP_;             // 2048 x 64 (pos,dep)
  const bf16_t* Vt = VcatT + bh * DEP_ * NS;             // 64 x 2048 (dep,pos)

  // ---- Stage Q tile (2KB) into LDS with async copy (CDNA5 path) ----
  if (tid < 128) {
    const bf16_t* gsrc = Q + (size_t)tid * 8;            // 16B per lane
#if __has_builtin(__builtin_amdgcn_global_load_async_to_lds_b128)
    __builtin_amdgcn_global_load_async_to_lds_b128(
        (__attribute__((address_space(1))) v4i*)gsrc,
        (__attribute__((address_space(3))) v4i*)(qtile + (size_t)tid * 8),
        0, 0);
#else
    *reinterpret_cast<uint4*>(qtile + (size_t)tid * 8) =
        *reinterpret_cast<const uint4*>(gsrc);
#endif
  }
#if __has_builtin(__builtin_amdgcn_s_wait_asynccnt)
  __builtin_amdgcn_s_wait_asynccnt(0);
#else
  asm volatile("s_wait_asynccnt 0x0" ::: "memory");
#endif
  __syncthreads();

  // ---- Phase 1: logits = (Q Kcat^T) * 0.125, causal mask (j <= i + PAST) ----
  v16bf a0 = load_frag_a(qtile, DEP_);
  v16bf a1 = load_frag_a(qtile + 32, DEP_);
  const int nt0 = wave * 16;

  auto qk_step = [&](int nt, const v16bf& kf0, const v16bf& kf1) {
    v8f acc = {};
    acc = WMMA_BF16(a0, kf0, acc);
    acc = WMMA_BF16(a1, kf1, acc);
    const int j = nt * 16 + col;
#pragma unroll
    for (int e = 0; e < 8; ++e) {
      const int m = e + 8 * half;
      logits[m * NS + j] = (j <= q_base + m + PAST) ? acc[e] * 0.125f : -1.0e10f;
    }
  };

  v16bf kA0 = load_frag_bT(K + (size_t)nt0 * 16 * DEP_, DEP_);
  v16bf kA1 = load_frag_bT(K + (size_t)nt0 * 16 * DEP_ + 32, DEP_);
  v16bf kB0, kB1;
#pragma unroll 1
  for (int i = 0; i < 7; ++i) {                 // 2 nt-steps per iteration
    const int ntA = nt0 + 2 * i;
    kB0 = load_frag_bT(K + (size_t)(ntA + 1) * 16 * DEP_, DEP_);
    kB1 = load_frag_bT(K + (size_t)(ntA + 1) * 16 * DEP_ + 32, DEP_);
    qk_step(ntA, kA0, kA1);
    kA0 = load_frag_bT(K + (size_t)(ntA + 2) * 16 * DEP_, DEP_);
    kA1 = load_frag_bT(K + (size_t)(ntA + 2) * 16 * DEP_ + 32, DEP_);
    qk_step(ntA + 1, kB0, kB1);
  }
  kB0 = load_frag_bT(K + (size_t)(nt0 + 15) * 16 * DEP_, DEP_);
  kB1 = load_frag_bT(K + (size_t)(nt0 + 15) * 16 * DEP_ + 32, DEP_);
  qk_step(nt0 + 14, kA0, kA1);
  qk_step(nt0 + 15, kB0, kB1);
  __syncthreads();

  // ---- Phase 2: row softmax (16 threads per row) ----
  const int row = tid >> 4, c = tid & 15;
  float mx = -3.4e38f;
  for (int j = c; j < NS; j += 16) mx = fmaxf(mx, logits[row * NS + j]);
  red[row * 16 + c] = mx;
  __syncthreads();
  if (c == 0) {
    float m2 = red[row * 16];
    for (int i = 1; i < 16; ++i) m2 = fmaxf(m2, red[row * 16 + i]);
    rstat[row] = m2;
  }
  __syncthreads();
  const float rm = rstat[row];
  float sum = 0.f;
  for (int j = c; j < NS; j += 16) {
    float e = __expf(logits[row * NS + j] - rm);
    logits[row * NS + j] = e;
    sum += e;
  }
  red[row * 16 + c] = sum;
  __syncthreads();
  if (c == 0) {
    float s2 = 0.f;
    for (int i = 0; i < 16; ++i) s2 += red[row * 16 + i];
    rstat[16 + row] = s2;
  }
  __syncthreads();
  const float inv = 1.0f / rstat[16 + row];
  float* awRow = attw + (bh * S_ + q_base + row) * NS;
  for (int j = c; j < NS; j += 16) {
    float p = logits[row * NS + j] * inv;
    awRow[j] = p;                      // fp32 attention_weights output
    probs[row * NS + j] = (bf16_t)p;   // bf16 for the AV WMMA
  }
  __syncthreads();

  // ---- Phase 3: out = P @ V. K (=2048) split across 8 waves, then reduce ----
  v8f oacc[4] = {};
  const int kb = wave * 256;

  auto av_load = [&](int kn, v16bf& apf, v16bf* bvf) {
    apf = load_frag_a(probs + kn, NS);
#pragma unroll
    for (int t = 0; t < 4; ++t)
      bvf[t] = load_frag_bT(Vt + (size_t)(t * 16) * NS + kn, NS);
  };
  auto av_compute = [&](const v16bf& apf, const v16bf* bvf) {
#pragma unroll
    for (int t = 0; t < 4; ++t) oacc[t] = WMMA_BF16(apf, bvf[t], oacc[t]);
  };

  v16bf apA, apB, bvA[4], bvB[4];
  av_load(kb, apA, bvA);
#pragma unroll 1
  for (int ks = 0; ks < 3; ++ks) {              // 2 k-steps per iteration
    av_load(kb + (2 * ks + 1) * 32, apB, bvB);
    av_compute(apA, bvA);
    av_load(kb + (2 * ks + 2) * 32, apA, bvA);
    av_compute(apB, bvB);
  }
  av_load(kb + 7 * 32, apB, bvB);
  av_compute(apA, bvA);
  av_compute(apB, bvB);

  float* part = logits;  // reuse: 8 waves * 16 * 64 f32 = 32KB
#pragma unroll
  for (int t = 0; t < 4; ++t)
#pragma unroll
    for (int e = 0; e < 8; ++e)
      part[((size_t)wave * 16 + e + 8 * half) * 64 + t * 16 + col] = oacc[t][e];
  __syncthreads();
  for (int o = tid; o < 16 * DEP_; o += 256) {
    const int m = o >> 6, d = o & 63;
    float s = 0.f;
#pragma unroll
    for (int w = 0; w < 8; ++w) s += part[(w * 16 + m) * 64 + d];
    AOut[((size_t)b * S_ + q_base + m) * D_ + h * DEP_ + d] = (bf16_t)s;  // (B,S,D)
  }
}

// ---------------------------------------------------------------------------
// Kernel 3: output projection out = AOut @ W2 + b2 (fp32 out), 32x64/wave,
// ping-pong double-buffered like qkv_gemm.
// ---------------------------------------------------------------------------

__global__ __launch_bounds__(32) void out_gemm(
    const bf16_t* __restrict__ A, const bf16_t* __restrict__ W2T,
    const float* __restrict__ b2, float* __restrict__ out) {
  const int n_base = blockIdx.x * 64;
  const int m_base = blockIdx.y * 32;
  const int lane = threadIdx.x & 31;
  const int half = lane >> 4, col = lane & 15;

  const bf16_t* Ar0 = A + (size_t)m_base * D_;
  const bf16_t* Ar1 = A + (size_t)(m_base + 16) * D_;
  const bf16_t* Bb  = W2T + (size_t)n_base * D_;

  v8f acc[2][4] = {};

  auto load_set = [&](int kn, v16bf& x0, v16bf& x1, v16bf* bb) {
    x0 = load_frag_a(Ar0 + kn, D_);
    x1 = load_frag_a(Ar1 + kn, D_);
#pragma unroll
    for (int t = 0; t < 4; ++t)
      bb[t] = load_frag_bT(Bb + (size_t)t * 16 * D_ + kn, D_);
  };
  auto compute = [&](const v16bf& x0, const v16bf& x1, const v16bf* bb) {
#pragma unroll
    for (int t = 0; t < 4; ++t) {
      acc[0][t] = WMMA_BF16(x0, bb[t], acc[0][t]);
      acc[1][t] = WMMA_BF16(x1, bb[t], acc[1][t]);
    }
  };

  v16bf a0A, a1A, bA[4], a0B, a1B, bB[4];
  load_set(0, a0A, a1A, bA);
#pragma unroll 1
  for (int ki = 0; ki < 15; ++ki) {
    load_set((2 * ki + 1) * 32, a0B, a1B, bB);
    compute(a0A, a1A, bA);
    load_set((2 * ki + 2) * 32, a0A, a1A, bA);
    compute(a0B, a1B, bB);
  }
  load_set(31 * 32, a0B, a1B, bB);
  compute(a0A, a1A, bA);
  compute(a0B, a1B, bB);

#pragma unroll
  for (int r = 0; r < 2; ++r) {
#pragma unroll
    for (int t = 0; t < 4; ++t) {
      const int n = n_base + 16 * t + col;
      const float bias = b2[n];
#pragma unroll
      for (int e = 0; e < 8; ++e) {
        const int m = m_base + 16 * r + e + 8 * half;
        out[(size_t)m * D_ + n] = acc[r][t][e] + bias;
      }
    }
  }
}

// ---------------------------------------------------------------------------
// Launch
// ---------------------------------------------------------------------------

extern "C" void kernel_launch(void* const* d_in, const int* in_sizes, int n_in,
                              void* d_out, int out_size, void* d_ws, size_t ws_size,
                              hipStream_t stream) {
  (void)in_sizes; (void)n_in; (void)out_size; (void)ws_size;

  const float* X    = (const float*)d_in[0];
  const float* past = (const float*)d_in[1];
  const float* W1   = (const float*)d_in[2];
  const float* b1   = (const float*)d_in[3];
  const float* W2   = (const float*)d_in[4];
  const float* b2   = (const float*)d_in[5];

  float* out     = (float*)d_out;                                   // (B,S,D)
  float* present = out + (size_t)B_ * S_ * D_;                      // (B,2,H,S,dep)
  float* attw    = present + (size_t)B_ * 2 * H_ * S_ * DEP_;       // (B,H,S,NS)

  // Workspace carve (all bf16 staging buffers), 256B-aligned.
  char* w = (char*)d_ws;
  size_t off = 0;
  auto carve = [&](size_t bytes) {
    void* p = w + off;
    off += (bytes + 255) & ~(size_t)255;
    return p;
  };
  bf16_t* Xbf   = (bf16_t*)carve((size_t)M_ * D_ * 2);               // 8 MB
  bf16_t* W1T   = (bf16_t*)carve((size_t)N1_ * D_ * 2);              // 6 MB  (N x K)
  bf16_t* W2T   = (bf16_t*)carve((size_t)D_ * D_ * 2);               // 2 MB  (N x K)
  bf16_t* Qbf   = (bf16_t*)carve((size_t)B_ * H_ * S_ * DEP_ * 2);   // 8 MB
  bf16_t* Kcat  = (bf16_t*)carve((size_t)B_ * H_ * NS * DEP_ * 2);   // 16 MB
  bf16_t* VcatT = (bf16_t*)carve((size_t)B_ * H_ * DEP_ * NS * 2);   // 16 MB
  bf16_t* AOut  = (bf16_t*)carve((size_t)M_ * D_ * 2);               // 8 MB

  // Prep: conversions / transposes / KV-cache fill.
  cvt_f32_bf16<<<(M_ * D_) / 256, 256, 0, stream>>>(X, Xbf);
  transpose_bf16<<<dim3(N1_ / 16, D_ / 16), dim3(16, 16), 0, stream>>>(W1, W1T, D_, N1_);
  transpose_bf16<<<dim3(D_ / 16, D_ / 16), dim3(16, 16), 0, stream>>>(W2, W2T, D_, D_);
  past_to_cache<<<(B_ * 2 * H_ * PAST * DEP_) / 256, 256, 0, stream>>>(past, Kcat, VcatT);

  // QKV projection (+present, +bf16 caches).
  qkv_gemm<<<dim3(N1_ / 64, M_ / 32), 32, 0, stream>>>(Xbf, W1T, b1, Qbf, Kcat, VcatT, present);

  // Fused attention (logits -> mask -> softmax -> weights out -> P@V).
  attention_kernel<<<dim3(S_ / 16, H_, B_), 256, SM_TOTAL, stream>>>(Qbf, Kcat, VcatT, attw, AOut);

  // Output projection.
  out_gemm<<<dim3(D_ / 64, M_ / 32), 32, 0, stream>>>(AOut, W2T, b2, out);
}
